// ParallelAttentionLayer_47536698032945
// MI455X (gfx1250) — compile-verified, matrix operand order checked by reading, not run on hardware
//
#include <hip/hip_runtime.h>
#include <hip/hip_bf16.h>

#define USE_TDM 1   // flip to 0 to fall back to the direct-global-load GEMM

// ---------------------------------------------------------------------------
// Types for CDNA5 WMMA (wave32): 16x16x32 f16 -> f32
// ---------------------------------------------------------------------------
typedef _Float16 h8   __attribute__((ext_vector_type(8)));
typedef _Float16 v16h __attribute__((ext_vector_type(16)));
typedef float    v8f  __attribute__((ext_vector_type(8)));
typedef unsigned v4u  __attribute__((ext_vector_type(4)));
typedef unsigned v8u  __attribute__((ext_vector_type(8)));

#define D_MODEL 512
#define D_INNER 1024
#define D_STATE 16
#define DT_RANK 32

// ---------------------------------------------------------------------------
// fp32 -> f16 conversion
// ---------------------------------------------------------------------------
__global__ void k_cvt_f16(const float* __restrict__ src, _Float16* __restrict__ dst, long n) {
    long i = (long)blockIdx.x * 256 + threadIdx.x;
    if (i < n) dst[i] = (_Float16)src[i];
}

// xdb[:, :32] (stride 64) -> packed f16 [M,32]
__global__ void k_cvt_dtrank(const float* __restrict__ src, _Float16* __restrict__ dst, long n) {
    long i = (long)blockIdx.x * 256 + threadIdx.x;
    if (i >= n) return;
    long m = i >> 5; int c = (int)(i & 31);
    dst[i] = (_Float16)src[m * 64 + c];
}

// ---------------------------------------------------------------------------
// Fragment load (ISA 7.12.2, 16-bit A/B 16x32 packing):
//   lane = (row%16) + 16*khalf ; v16h = {row[kk+khalf*8 .. +7], row[kk+16+khalf*8 .. +7]}
// Works for both the A operand (rows of A) and the B operand (rows of W, since
// C = A * W^T means B's per-lane column n == W row n).
// ---------------------------------------------------------------------------
__device__ __forceinline__ v16h load_frag(const _Float16* p, int khalf) {
    union { v16h v; h8 h[2]; } u;
    u.h[0] = *(const h8*)(p + khalf * 8);
    u.h[1] = *(const h8*)(p + 16 + khalf * 8);
    return u.v;
}

__device__ __forceinline__ void epilogue_store(float* C, const float* bias,
                                               v8f* accs, int EPI,
                                               int m0, int n0, int mr, int khalf, int N) {
    #pragma unroll
    for (int t = 0; t < 4; t++) {
        const int col = n0 + t * 16 + mr;
        #pragma unroll
        for (int r = 0; r < 8; r++) {
            const int row = m0 + r + 8 * khalf;   // C layout: VGPR r -> M=r / r+8 by lane half
            float v = accs[t][r];
            if (EPI == 1) {
                v += bias[col];
                v = (v > 20.f) ? v : log1pf(__expf(v));   // softplus
            }
            C[(long)row * N + col] = v;
        }
    }
}

// ---------------------------------------------------------------------------
// Fallback WMMA GEMM (direct global loads for both operands).
// C[M,N] f32 = A[M,K] f16 * W[N,K]^T f16. Block 128x64 tile, 8 waves.
// ---------------------------------------------------------------------------
template <int EPI>
__global__ void k_gemm(const _Float16* __restrict__ A, const _Float16* __restrict__ W,
                       float* __restrict__ C, const float* __restrict__ bias,
                       int M, int N, int K) {
    const int lane  = threadIdx.x & 31;
    const int wave  = threadIdx.x >> 5;
    const int mr    = lane & 15;
    const int khalf = lane >> 4;
    const int m0    = blockIdx.y * 128 + wave * 16;
    const int n0    = blockIdx.x * 64;

    const _Float16* aRow = A + (long)(m0 + mr) * K;
    const _Float16* wRow = W + (long)(n0 + mr) * K;

    v8f accs[4];
    #pragma unroll
    for (int t = 0; t < 4; t++)
        for (int i = 0; i < 8; i++) accs[t][i] = 0.f;

    for (int kk = 0; kk < K; kk += 32) {
        __builtin_prefetch(aRow + kk + 128, 0, 1);
        v16h a = load_frag(aRow + kk, khalf);
        #pragma unroll
        for (int t = 0; t < 4; t++) {
            v16h b = load_frag(wRow + (long)t * 16 * K + kk, khalf);
            accs[t] = __builtin_amdgcn_wmma_f32_16x16x32_f16(false, a, false, b, (short)0, accs[t], false, false);
        }
    }
    epilogue_store(C, bias, accs, EPI, m0, n0, mr, khalf, N);
}

#if USE_TDM
// ---------------------------------------------------------------------------
// TDM-fed WMMA GEMM: the 64-row W slab is DMA'd into LDS by the Tensor Data
// Mover (one tensor_load_to_lds per block per 128-half K-chunk, double
// buffered, TENSORcnt-tracked). TDM pad_enable inserts 4 dwords per row so
// the LDS row stride is (chunk*2+16) bytes == 4 banks mod 64: the 16 lanes'
// ds_load_b128 fragment reads hit disjoint bank quads.
// A stream stays on direct global_load_b128 + global_prefetch_b8.
// ---------------------------------------------------------------------------
__device__ __forceinline__ void tdm_issue_wslab(const _Float16* W, long rowBase, int K,
                                                int c, int chunk, int ivCode,
                                                unsigned ldsOff) {
    unsigned long long ga = (unsigned long long)(size_t)(const void*)(W + rowBase + (long)c * chunk);
    v4u g0; v8u g1;
    g0[0] = 1u;                                            // count=1, user mode
    g0[1] = ldsOff;                                        // lds_addr (dynamic LDS base = 0)
    g0[2] = (unsigned)(ga & 0xffffffffu);                  // global_addr[31:0]
    g0[3] = (unsigned)((ga >> 32) & 0x1ffffffu)            // global_addr[56:32]
          | 0x80000000u;                                   // type=2 ("image")
    g1[0] = (1u << 16)                                     // data_size = 2 bytes
          | (1u << 20)                                     // pad_enable
          | ((unsigned)ivCode << 22)                       // pad_interval = one row of dwords
          | (3u << 25);                                    // pad_amount = 4 dwords
    g1[1] = ((unsigned)chunk & 0xffffu) << 16;             // tensor_dim0[15:0]
    g1[2] = (((unsigned)chunk >> 16) & 0xffffu)            // tensor_dim0[31:16]
          | (64u << 16);                                   // tensor_dim1[15:0] = 64 rows
    g1[3] = ((unsigned)chunk << 16);                       // tensor_dim1[31:16]=0 | tile_dim0=chunk
    g1[4] = 64u;                                           // tile_dim1=64, tile_dim2=0
    g1[5] = (unsigned)K;                                   // tensor_dim0_stride[31:0] = K elements
    g1[6] = 0u;                                            // stride[47:32]=0, dim1_stride lo=0
    g1[7] = 0u;
    asm volatile("tensor_load_to_lds %0, %1" :: "s"(g0), "s"(g1) : "memory");
}

template <int EPI>
__global__ void k_gemm_tdm(const _Float16* __restrict__ A, const _Float16* __restrict__ W,
                           float* __restrict__ C, const float* __restrict__ bias,
                           int M, int N, int K) {
    extern __shared__ char smem[];
    const int lane  = threadIdx.x & 31;
    const int wave  = threadIdx.x >> 5;
    const int mr    = lane & 15;
    const int khalf = lane >> 4;
    const int m0    = blockIdx.y * 128 + wave * 16;
    const int n0    = blockIdx.x * 64;

    const int chunk     = (K < 128) ? K : 128;     // halfs per K-chunk
    const int rowStride = chunk * 2 + 16;          // bytes, incl. TDM pad (4 dwords)
    const int bufBytes  = 64 * rowStride;
    const int nChunks   = K / chunk;
    const int ivCode    = (chunk == 128) ? 5 : ((chunk == 64) ? 4 : 3); // log2(chunk/2)-1

    const _Float16* aRow = A + (long)(m0 + mr) * K;
    const long wRowBase  = (long)n0 * K;

    v8f accs[4];
    #pragma unroll
    for (int t = 0; t < 4; t++)
        for (int i = 0; i < 8; i++) accs[t][i] = 0.f;

    if (wave == 0)
        tdm_issue_wslab(W, wRowBase, K, 0, chunk, ivCode, 0u);

    int kk = 0;
    for (int c = 0; c < nChunks; ++c) {
        if (wave == 0) {
            if (c + 1 < nChunks) {
                tdm_issue_wslab(W, wRowBase, K, c + 1, chunk, ivCode,
                                (unsigned)(((c + 1) & 1) * bufBytes));
                __builtin_amdgcn_s_wait_tensorcnt(1);   // chunk c landed
            } else {
                __builtin_amdgcn_s_wait_tensorcnt(0);
            }
        }
        __syncthreads();                                 // publish LDS tile to all waves

        const char* wbuf = smem + (c & 1) * bufBytes;
        for (int ks = 0; ks < chunk; ks += 32) {
            __builtin_prefetch(aRow + kk + 128, 0, 1);
            v16h a = load_frag(aRow + kk, khalf);
            #pragma unroll
            for (int t = 0; t < 4; t++) {
                const _Float16* lrow = (const _Float16*)(wbuf + (t * 16 + mr) * rowStride);
                v16h b = load_frag(lrow + ks, khalf);
                accs[t] = __builtin_amdgcn_wmma_f32_16x16x32_f16(false, a, false, b, (short)0, accs[t], false, false);
            }
            kk += 32;
        }
        __syncthreads();                                 // done reading before buffer reuse
    }
    epilogue_store(C, bias, accs, EPI, m0, n0, mr, khalf, N);
}
#endif // USE_TDM

// ---------------------------------------------------------------------------
// Depthwise causal conv(4) + SiLU. Reads xi = xz[:, :1024]; writes xc (f32) + xch (f16).
// ---------------------------------------------------------------------------
__global__ void k_conv_silu(const float* __restrict__ xz, const float* __restrict__ cw,
                            const float* __restrict__ cb, float* __restrict__ xc,
                            _Float16* __restrict__ xch, int L, long total) {
    long i = (long)blockIdx.x * 256 + threadIdx.x;
    if (i >= total) return;
    int e = (int)(i & (D_INNER - 1));
    long row = i >> 10;                 // b*L + l
    int l = (int)(row % L);
    float acc = cb[e];
    #pragma unroll
    for (int k = 0; k < 4; k++) {
        int ll = l + k - 3;
        if (ll >= 0) acc += xz[(row + (ll - l)) * 2048 + e] * cw[e * 4 + k];
    }
    float s = acc / (1.f + __expf(-acc));
    xc[i] = s;
    xch[i] = (_Float16)s;
}

// ---------------------------------------------------------------------------
// Selective scan: one thread per (b, e) channel, 16-wide state in registers.
// B_t/C_t staged through LDS. Fuses D-skip + SiLU(z) gating; writes y as f16.
// ---------------------------------------------------------------------------
__global__ void k_scan(const float* __restrict__ dt, const float* __restrict__ xc,
                       const float* __restrict__ xz, const float* __restrict__ xdb,
                       const float* __restrict__ A_log, const float* __restrict__ Dp,
                       _Float16* __restrict__ yh, int L) {
    const int b = blockIdx.x >> 2;
    const int e = ((blockIdx.x & 3) << 8) + threadIdx.x;
    float a[D_STATE], h[D_STATE];
    #pragma unroll
    for (int n = 0; n < D_STATE; n++) {
        a[n] = -__expf(A_log[e * D_STATE + n]);
        h[n] = 0.f;
    }
    const float dval = Dp[e];
    __shared__ float bc[32];
    for (int l = 0; l < L; l++) {
        const long m = (long)b * L + l;
        if (threadIdx.x < 32) bc[threadIdx.x] = xdb[m * 64 + DT_RANK + threadIdx.x];
        __syncthreads();
        const float dtv = dt[m * D_INNER + e];
        const float xv  = xc[m * D_INNER + e];
        float y = 0.f;
        #pragma unroll
        for (int n = 0; n < D_STATE; n++) {
            float dA = __expf(dtv * a[n]);
            h[n] = dA * h[n] + (dtv * xv) * bc[n];
            y += h[n] * bc[D_STATE + n];
        }
        const float zv = xz[m * 2048 + D_INNER + e];
        const float yt = (y + dval * xv) * (zv / (1.f + __expf(-zv)));
        yh[m * D_INNER + e] = (_Float16)yt;
        __syncthreads();
    }
}

// ---------------------------------------------------------------------------
// out = LayerNorm(resid + mm) over D_MODEL=512. One block (256 threads) per row.
// ---------------------------------------------------------------------------
__global__ void k_resid_ln(const float* __restrict__ resid, const float* __restrict__ mm,
                           const float* __restrict__ g, const float* __restrict__ bb,
                           float* __restrict__ out) {
    const long m = blockIdx.x;
    const int t = threadIdx.x;
    const float v0 = resid[m * 512 + t]       + mm[m * 512 + t];
    const float v1 = resid[m * 512 + 256 + t] + mm[m * 512 + 256 + t];
    __shared__ float ssum[256], ssq[256];
    ssum[t] = v0 + v1;
    ssq[t]  = v0 * v0 + v1 * v1;
    __syncthreads();
    for (int s = 128; s > 0; s >>= 1) {
        if (t < s) { ssum[t] += ssum[t + s]; ssq[t] += ssq[t + s]; }
        __syncthreads();
    }
    const float mean = ssum[0] * (1.f / 512.f);
    const float var  = ssq[0] * (1.f / 512.f) - mean * mean;
    const float inv  = rsqrtf(var + 1e-5f);
    out[m * 512 + t]       = (v0 - mean) * inv * g[t]       + bb[t];
    out[m * 512 + 256 + t] = (v1 - mean) * inv * g[256 + t] + bb[256 + t];
}

// ---------------------------------------------------------------------------
// Host-side orchestration
// ---------------------------------------------------------------------------
struct MambaW {
    const float *in_proj, *conv_w, *conv_b, *x_proj, *dt_proj, *dt_bias, *A_log, *D, *out_proj;
};

static inline void launch_gemm(int epi, dim3 grid, hipStream_t s,
                               const _Float16* A, const _Float16* W, float* C,
                               const float* bias, int M, int N, int K) {
#if USE_TDM
    const int chunk = (K < 128) ? K : 128;
    const size_t smem = 2u * 64u * (size_t)(chunk * 2 + 16);
    if (epi) k_gemm_tdm<1><<<grid, 256, smem, s>>>(A, W, C, bias, M, N, K);
    else     k_gemm_tdm<0><<<grid, 256, smem, s>>>(A, W, C, bias, M, N, K);
#else
    if (epi) k_gemm<1><<<grid, 256, 0, s>>>(A, W, C, bias, M, N, K);
    else     k_gemm<0><<<grid, 256, 0, s>>>(A, W, C, bias, M, N, K);
#endif
}

static void run_mamba(hipStream_t stream, char* ws,
                      const float* x_in, const float* ln_g, const float* ln_b,
                      float* dest, const MambaW& w, int B, int L) {
    const long M = (long)B * L;
    size_t off = 0;
    auto bump = [&](size_t bytes) { size_t o = off; off += (bytes + 255) & ~(size_t)255; return o; };
    _Float16* WIH  = (_Float16*)(ws + bump(2048L * 512 * 2));
    _Float16* WXH  = (_Float16*)(ws + bump(64L * 1024 * 2));
    _Float16* WDH  = (_Float16*)(ws + bump(1024L * 32 * 2));
    _Float16* WOH  = (_Float16*)(ws + bump(512L * 1024 * 2));
    _Float16* XH   = (_Float16*)(ws + bump(4096L * 512 * 2));
    float*    XZ   = (float*)   (ws + bump(4096L * 2048 * 4));
    float*    XC   = (float*)   (ws + bump(4096L * 1024 * 4));
    _Float16* XCH  = (_Float16*)(ws + bump(4096L * 1024 * 2));
    float*    XDB  = (float*)   (ws + bump(4096L * 64 * 4));
    _Float16* XDBH = (_Float16*)(ws + bump(4096L * 32 * 2));
    float*    DT   = (float*)   (ws + bump(4096L * 1024 * 4));
    _Float16* YH   = (_Float16*)(ws + bump(4096L * 1024 * 2));
    float*    GOUT = XZ;   // xz is dead after the scan; reuse for out_proj result

    auto grid1d = [](long n) { return dim3((unsigned)((n + 255) / 256)); };

    // weights -> f16
    k_cvt_f16<<<grid1d(2048L * 512), 256, 0, stream>>>(w.in_proj,  WIH, 2048L * 512);
    k_cvt_f16<<<grid1d(64L * 1024),  256, 0, stream>>>(w.x_proj,   WXH, 64L * 1024);
    k_cvt_f16<<<grid1d(1024L * 32),  256, 0, stream>>>(w.dt_proj,  WDH, 1024L * 32);
    k_cvt_f16<<<grid1d(512L * 1024), 256, 0, stream>>>(w.out_proj, WOH, 512L * 1024);
    // activations -> f16
    k_cvt_f16<<<grid1d(M * 512), 256, 0, stream>>>(x_in, XH, M * 512);

    // G1: xz[M,2048] = x @ in_proj^T        (K=512)
    launch_gemm(0, dim3(2048 / 64, (unsigned)(M / 128)), stream, XH, WIH, XZ, nullptr, (int)M, 2048, 512);
    // conv(4) + SiLU
    k_conv_silu<<<grid1d(M * 1024), 256, 0, stream>>>(XZ, w.conv_w, w.conv_b, XC, XCH, L, M * 1024);
    // G2: xdb[M,64] = xc @ x_proj^T          (K=1024)
    launch_gemm(0, dim3(1, (unsigned)(M / 128)), stream, XCH, WXH, XDB, nullptr, (int)M, 64, 1024);
    // pack dt-rank slice to f16
    k_cvt_dtrank<<<grid1d(M * 32), 256, 0, stream>>>(XDB, XDBH, M * 32);
    // G3: dt[M,1024] = softplus(xdb[:, :32] @ dt_proj^T + dt_bias)   (K=32)
    launch_gemm(1, dim3(1024 / 64, (unsigned)(M / 128)), stream, XDBH, WDH, DT, w.dt_bias, (int)M, 1024, 32);
    // selective scan (fuses D-skip + SiLU(z) gating), writes y as f16
    k_scan<<<B * 4, 256, 0, stream>>>(DT, XC, XZ, XDB, w.A_log, w.D, YH, L);
    // G4: gout[M,512] = y @ out_proj^T       (K=1024)
    launch_gemm(0, dim3(512 / 64, (unsigned)(M / 128)), stream, YH, WOH, GOUT, nullptr, (int)M, 512, 1024);
    // residual + LayerNorm
    k_resid_ln<<<(unsigned)M, 256, 0, stream>>>(x_in, GOUT, ln_g, ln_b, dest);
}

extern "C" void kernel_launch(void* const* d_in, const int* in_sizes, int n_in,
                              void* d_out, int out_size, void* d_ws, size_t ws_size,
                              hipStream_t stream) {
    (void)in_sizes; (void)n_in; (void)out_size; (void)ws_size;
    auto f = [&](int i) { return (const float*)d_in[i]; };
    auto mk = [&](int b) {
        MambaW w;
        w.in_proj = f(b + 0); w.conv_w = f(b + 1); w.conv_b = f(b + 2);
        w.x_proj  = f(b + 3); w.dt_proj = f(b + 4); w.dt_bias = f(b + 5);
        w.A_log   = f(b + 6); w.D = f(b + 7); w.out_proj = f(b + 8);
        return w;
    };
    // dict order: seq1, mask1, seq2, mask2, p12(9), p21(9), p1(9), p2(9),
    //             ln12(g,b), ln21(g,b), ln1(g,b), ln2(g,b)
    const float* seq1 = f(0);
    const float* seq2 = f(2);
    MambaW p12 = mk(4), p21 = mk(13), p1 = mk(22), p2 = mk(31);
    const float *ln12g = f(40), *ln12b = f(41), *ln21g = f(42), *ln21b = f(43);
    const float *ln1g  = f(44), *ln1b  = f(45), *ln2g  = f(46), *ln2b  = f(47);

    char* ws = (char*)d_ws;
    const size_t SCRATCH = 93000000;                  // beyond bump-allocated region
    float* sb1 = (float*)(ws + SCRATCH);              // 4*1024*512 f32 = 8 MB
    float* sb2 = (float*)(ws + SCRATCH + 8388608);    // 4*512*512  f32 = 4 MB

    float* out1 = (float*)d_out;                      // seq1 result: 4*1024*512
    float* out2 = out1 + 4L * 1024 * 512;             // seq2 result: 4*512*512

    run_mamba(stream, ws, seq1, ln12g, ln12b, sb1,  p12, 4, 1024);
    run_mamba(stream, ws, seq2, ln21g, ln21b, sb2,  p21, 4, 512);
    run_mamba(stream, ws, sb1,  ln1g,  ln1b,  out1, p1,  4, 1024);
    run_mamba(stream, ws, sb2,  ln2g,  ln2b,  out2, p2,  4, 512);
}
